// FusedLinearCrossEntropyLoss_82978768159371
// MI455X (gfx1250) — matrix operand order repaired and checked by reading, not compiled, not run on hardware
//
#include <hip/hip_runtime.h>
#include <hip/hip_bf16.h>

#define BT 4096
#define H  2048
#define V  32000
#define IGNORE_INDEX (-100)

#define M_WG   32            // tokens per workgroup (A rows resident in LDS)
#define N_WG   64            // vocab columns in flight per workgroup
#define KC     128           // K-chunk per LDS B-buffer
#define NKC    (H / KC)      // 16 K-chunks per column block
#define VSPLIT 4             // vocab split across blockIdx.y for occupancy
#define VS     (V / VSPLIT)  // 8000 columns per split

typedef __attribute__((ext_vector_type(16))) __bf16 v16bf;
typedef __attribute__((ext_vector_type(8)))  __bf16 v8bf;
typedef __attribute__((ext_vector_type(8)))  float  v8f;

union ABfrag { v16bf v; v8bf h[2]; };

// XOR swizzle on 16B chunks: logical chunk c of row/col r -> physical slot.
// A 16-lane fragment read (distinct r&15, same c) maps to 16 distinct slots,
// so the 16x16B requests tile all 64 LDS banks exactly once: conflict-free
// with zero padding (we have zero LDS slack: 128K + 32K = 160K = the cap).
__device__ __forceinline__ int swz(int c, int r) {
  return (c & ~15) | ((c ^ r) & 15);
}

// Cooperative stage of one weight chunk [N_WG cols x KC k] f32 -> bf16 into
// swizzled sB. 4 threads per column, 4 chunks (32 K) each; coalesced float4.
__device__ __forceinline__ void stage_b(__bf16* __restrict__ dst,
                                        const float* __restrict__ weight,
                                        int v0, int kb, int scol, int sc4) {
  const float4* s4 = (const float4*)&weight[(size_t)(v0 + scol) * H + kb + sc4 * 8];
  __bf16* d = dst + scol * KC;
  #pragma unroll
  for (int j = 0; j < 4; ++j) {
    float4 a = s4[2 * j], b = s4[2 * j + 1];
    v8bf pk = {(__bf16)a.x, (__bf16)a.y, (__bf16)a.z, (__bf16)a.w,
               (__bf16)b.x, (__bf16)b.y, (__bf16)b.z, (__bf16)b.w};
    *(v8bf*)(d + swz(sc4 + j, scol & 15) * 8) = pk;   // 16B swizzled store
  }
}

// WG = 32 tokens x full vocab slice. 8 waves = 2 row-tiles x 4 col-tiles.
// sA: 32x2048 bf16 x-tile (128 KB), staged once, XOR-swizzled by row.
// sB: double-buffered 64x128 bf16 weight chunk (2x16 KB), XOR-swizzled by col,
//     converted once per WG and shared by all waves.
// Inner loop: 4 conflict-free ds_load_b128 + 1 v_wmma_f32_16x16x32_bf16.
__global__ __launch_bounds__(256, 1) void flce_main(
    const float* __restrict__ x, const int* __restrict__ target,
    const float* __restrict__ weight, const float* __restrict__ bias,
    float* __restrict__ pMax, float* __restrict__ pSum, float* __restrict__ pTgt)
{
  __shared__ __attribute__((aligned(16))) __bf16 sA[M_WG * H];       // 128 KB
  __shared__ __attribute__((aligned(16))) __bf16 sB[2 * N_WG * KC];  //  32 KB

  const int tid   = threadIdx.x;
  const int wave  = tid >> 5;
  const int lane  = tid & 31;
  const int nlane = lane & 15;     // WMMA N column / A M row within tile
  const int khalf = lane >> 4;     // lane half selects K sub-range
  const int mt    = wave & 1;      // row-tile 0/1
  const int ng    = wave >> 1;     // column-tile 0..3
  const int m0    = blockIdx.x * M_WG;
  const int vbase = blockIdx.y * VS;

  // ---- stage x tile (f32 -> bf16) into swizzled LDS, 16B granules ----
  for (int cidx = tid; cidx < M_WG * (H / 8); cidx += 256) {
    const int row = cidx >> 8;          // H/8 = 256 chunks per row
    const int c   = cidx & 255;
    const float4* s4 = (const float4*)&x[(size_t)(m0 + row) * H + c * 8];
    float4 a = s4[0], b = s4[1];
    v8bf pk = {(__bf16)a.x, (__bf16)a.y, (__bf16)a.z, (__bf16)a.w,
               (__bf16)b.x, (__bf16)b.y, (__bf16)b.z, (__bf16)b.w};
    *(v8bf*)(&sA[row * H] + swz(c, row & 15) * 8) = pk;
  }

  int tgts[8];
  #pragma unroll
  for (int i = 0; i < 8; ++i) tgts[i] = target[m0 + mt * 16 + khalf * 8 + i];

  float rmax[8], rsum[8], tcap[8];
  #pragma unroll
  for (int i = 0; i < 8; ++i) {
    rmax[i] = -__builtin_inff(); rsum[i] = 0.f; tcap[i] = 0.f;
  }

  const __bf16* aRow = &sA[(mt * 16 + nlane) * H];
  const int scol = tid >> 2;           // staging column 0..63
  const int sc4  = (tid & 3) * 4;      // staging first chunk (of 16) in column

  for (int v0 = vbase; v0 < vbase + VS; v0 += N_WG) {
    stage_b(&sB[0], weight, v0, 0, scol, sc4);    // prime buffer 0
    v8f c = {0.f, 0.f, 0.f, 0.f, 0.f, 0.f, 0.f, 0.f};

    for (int kc = 0; kc < NKC; ++kc) {
      __syncthreads();                            // buffer kc&1 is ready
      if (kc + 1 < NKC)                           // overlap: stage next chunk
        stage_b(&sB[((kc + 1) & 1) * N_WG * KC], weight, v0, (kc + 1) * KC,
                scol, sc4);
      const __bf16* bCol = &sB[((kc & 1) * N_WG + ng * 16 + nlane) * KC];
      #pragma unroll
      for (int kk = 0; kk < KC; kk += 32) {
        ABfrag a, b;
        // A 16x32 bf16 frag: 16B chunks at K = kc*128+kk+khalf*8 and +16
        const int ca = kc * 16 + (kk >> 3) + khalf;
        a.h[0] = *(const v8bf*)(aRow + swz(ca,     nlane) * 8);
        a.h[1] = *(const v8bf*)(aRow + swz(ca + 2, nlane) * 8);
        // B 32x16 bf16 frag: 16B chunks at K = kk+khalf*8 and +16, column = nlane
        const int cb = (kk >> 3) + khalf;
        b.h[0] = *(const v8bf*)(bCol + ((cb       ^ nlane) * 8));
        b.h[1] = *(const v8bf*)(bCol + (((cb + 2) ^ nlane) * 8));
        c = __builtin_amdgcn_wmma_f32_16x16x32_bf16(
                false, a.v, false, b.v, (short)0, c, false, false);
      }
    }

    const float bv  = bias[v0 + ng * 16 + nlane];
    const int  vcol = v0 + ng * 16 + nlane;
    #pragma unroll
    for (int i = 0; i < 8; ++i) {       // C slot i: row mt*16+khalf*8+i, col nlane
      float l = c[i] + bv;
      if (tgts[i] == vcol) tcap[i] = l;            // target logit capture
      float nm = fmaxf(rmax[i], l);                // online logsumexp
      rsum[i] = rsum[i] * __expf(rmax[i] - nm) + __expf(l - nm);
      rmax[i] = nm;
    }
  }

  // ---- combine across the 16 columns held by each lane half ----
  #pragma unroll
  for (int i = 0; i < 8; ++i) {
    #pragma unroll
    for (int off = 1; off < 16; off <<= 1) {
      float om = __shfl_xor(rmax[i], off, 32);
      float os = __shfl_xor(rsum[i], off, 32);
      float ot = __shfl_xor(tcap[i], off, 32);
      float nm = fmaxf(rmax[i], om);
      rsum[i] = rsum[i] * __expf(rmax[i] - nm) + os * __expf(om - nm);
      rmax[i] = nm;
      tcap[i] += ot;
    }
  }

  // ---- combine across the 4 column-groups via LDS (overlays sB) ----
  float* sRed = (float*)sB;            // 384 floats; sB no longer needed
  __syncthreads();                      // all waves done reading sB
  if (nlane == 0) {
    #pragma unroll
    for (int i = 0; i < 8; ++i) {
      int row = mt * 16 + khalf * 8 + i;
      sRed[0 * M_WG * 4 + row * 4 + ng] = rmax[i];
      sRed[1 * M_WG * 4 + row * 4 + ng] = rsum[i];
      sRed[2 * M_WG * 4 + row * 4 + ng] = tcap[i];
    }
  }
  __syncthreads();
  if (tid < M_WG) {
    float M = -__builtin_inff(), S = 0.f, T = 0.f;
    #pragma unroll
    for (int g = 0; g < 4; ++g) {
      float gm = sRed[0 * M_WG * 4 + tid * 4 + g];
      float nm = fmaxf(M, gm);
      S = S * __expf(M - nm) + sRed[1 * M_WG * 4 + tid * 4 + g] * __expf(gm - nm);
      M = nm;
      T += sRed[2 * M_WG * 4 + tid * 4 + g];
    }
    const int t = m0 + tid;
    pMax[blockIdx.y * BT + t] = M;
    pSum[blockIdx.y * BT + t] = S;
    pTgt[blockIdx.y * BT + t] = T;
  }
}

// Merge VSPLIT partial logsumexp states per token, masked mean -> scalar loss.
__global__ __launch_bounds__(256) void flce_reduce(
    const float* __restrict__ pMax, const float* __restrict__ pSum,
    const float* __restrict__ pTgt, const int* __restrict__ target,
    float* __restrict__ out)
{
  float s = 0.f, cnt = 0.f;
  for (int t = threadIdx.x; t < BT; t += 256) {
    int tg = target[t];
    if (tg != IGNORE_INDEX) {
      float M = -__builtin_inff(), S = 0.f, T = 0.f;
      #pragma unroll
      for (int p = 0; p < VSPLIT; ++p) {
        float gm = pMax[p * BT + t];
        float nm = fmaxf(M, gm);
        S = S * __expf(M - nm) + pSum[p * BT + t] * __expf(gm - nm);
        M = nm;
        T += pTgt[p * BT + t];
      }
      s += (M + __logf(S)) - T;         // lse - target_logit
      cnt += 1.f;
    }
  }
  #pragma unroll
  for (int off = 16; off > 0; off >>= 1) {
    s   += __shfl_down(s, off, 32);
    cnt += __shfl_down(cnt, off, 32);
  }
  __shared__ float ss[8], sc[8];
  const int wv = threadIdx.x >> 5;
  if ((threadIdx.x & 31) == 0) { ss[wv] = s; sc[wv] = cnt; }
  __syncthreads();
  if (threadIdx.x == 0) {
    float S = 0.f, C = 0.f;
    #pragma unroll
    for (int i = 0; i < 8; ++i) { S += ss[i]; C += sc[i]; }
    out[0] = S / fmaxf(C, 1.f);
  }
}

extern "C" void kernel_launch(void* const* d_in, const int* in_sizes, int n_in,
                              void* d_out, int out_size, void* d_ws, size_t ws_size,
                              hipStream_t stream) {
  const float* x      = (const float*)d_in[0];
  const int*   target = (const int*)d_in[1];   // int32 (JAX default x64 off)
  const float* weight = (const float*)d_in[2];
  const float* bias   = (const float*)d_in[3];

  float* pMax = (float*)d_ws;                  // [VSPLIT][BT]
  float* pSum = pMax + VSPLIT * BT;
  float* pTgt = pSum + VSPLIT * BT;            // total 196 KB of scratch

  dim3 grid(BT / M_WG, VSPLIT);
  flce_main<<<grid, 256, 0, stream>>>(x, target, weight, bias, pMax, pSum, pTgt);
  flce_reduce<<<1, 256, 0, stream>>>(pMax, pSum, pTgt, target, (float*)d_out);
}
